// JSDMAttention_4303557231005
// MI455X (gfx1250) — matrix operand order repaired and checked by reading, not compiled, not run on hardware
//
#include <hip/hip_runtime.h>

// ---------------- problem constants ----------------
#define B_DIM    4
#define S_DIM    256
#define T_DIM    64
#define CST_DIM  64
#define CECO_DIM 32
#define H_DIM    512
#define HALF_H   4      // heads per sub-block
#define D_DIM    64     // head dim
#define AH_DIM   256    // sub-block all_head_size
#define FIREH    32
#define SCALE_QK 0.125f // 1/sqrt(64)
#define MAXSP    180.0f
#define MAXTM    365.0f

typedef __attribute__((ext_vector_type(8)))  __bf16 v8bf;
typedef __attribute__((ext_vector_type(16))) __bf16 v16bf;
typedef __attribute__((ext_vector_type(8)))  float  v8f;

// ---------------- WMMA helpers ----------------
__device__ __forceinline__ v8f wmma_bf16(v16bf a, v16bf b, v8f c) {
  // v_wmma_f32_16x16x32_bf16: D = A(16x32) * B(32x16) + C(16x16 f32)
  return __builtin_amdgcn_wmma_f32_16x16x32_bf16(false, a, false, b,
                                                 (short)0, c, false, false);
}

__device__ __forceinline__ v16bf join8(v8bf lo, v8bf hi) {
  return __builtin_shufflevector(lo, hi, 0,1,2,3,4,5,6,7,8,9,10,11,12,13,14,15);
}

// A-matrix 16x32 bf16 fragment, row-major source row pointer.
// lanes 0-15 hold K = k0+0..15 ; lanes 16-31 hold K = k0+16..31
__device__ __forceinline__ v16bf fragA_global(const __bf16* __restrict__ row, int k0, int hi) {
  const __bf16* p0 = row + k0 + hi * 16;
  const __bf16* p1 = row + k0 + hi * 16 + 8;
  return join8(*(const v8bf*)p0, *(const v8bf*)p1);
}

// B-matrix 32x16 bf16 fragment where B[k][n] = Wrow_n[k] (row-major, contiguous in k).
// lanes 0-15 hold K = k0+0..15 ; lanes 16-31 hold K = k0+16..31
__device__ __forceinline__ v16bf fragB_rows(const __bf16* __restrict__ rowN, int k0, int hi) {
  const __bf16* p = rowN + k0 + hi * 16;
  return join8(*(const v8bf*)p, *(const v8bf*)(p + 8));
}

// A fragment from an f32 LDS row (softmax probabilities), converted to bf16.
__device__ __forceinline__ v16bf fragA_lds(const float* __restrict__ row, int k0, int hi) {
  v16bf a;
  int b0 = k0 + hi * 16;
  int b1 = k0 + hi * 16 + 8;
#pragma unroll
  for (int i = 0; i < 8; ++i) a[i] = (__bf16)row[b0 + i];
#pragma unroll
  for (int i = 0; i < 8; ++i) a[8 + i] = (__bf16)row[b1 + i];
  return a;
}

// B fragment gathered column-wise from an LDS-staged V tile (row stride = 64 bf16).
__device__ __forceinline__ v16bf fragB_lds_col(const __bf16* __restrict__ Vs,
                                               int k0, int hi, int col) {
  v16bf b;
#pragma unroll
  for (int kk = 0; kk < 16; ++kk)
    b[kk] = Vs[(k0 + hi * 16 + kk) * D_DIM + col];
  return b;
}

// ---------------- f32 -> bf16 cast ----------------
__global__ void cast_bf16_k(const float* __restrict__ x, __bf16* __restrict__ y, long n) {
  long i = (long)blockIdx.x * blockDim.x + threadIdx.x;
  long stride = (long)gridDim.x * blockDim.x;
  for (; i < n; i += stride) y[i] = (__bf16)x[i];
}

// ---------------- FIRE distance bias ----------------
__device__ __forceinline__ float softplusf(float x) {
  return x > 20.f ? x : log1pf(__expf(x));
}

__global__ void fire_bias_k(const float* __restrict__ sd, const float* __restrict__ td,
                            const float* __restrict__ sp_logc, const float* __restrict__ sp_w1,
                            const float* __restrict__ sp_w2, const float* __restrict__ tm_logc,
                            const float* __restrict__ tm_w1, const float* __restrict__ tm_w2,
                            const float* __restrict__ sp_scale, const float* __restrict__ tm_scale,
                            float* __restrict__ out) {
  long idx = (long)blockIdx.x * blockDim.x + threadIdx.x;
  const long total = (long)B_DIM * S_DIM * T_DIM * CST_DIM;
  if (idx >= total) return;
  int c = idx % CST_DIM;
  int t = (idx / CST_DIM) % T_DIM;
  int s = (idx / (CST_DIM * T_DIM)) % S_DIM;
  int b = idx / ((long)CST_DIM * T_DIM * S_DIM);

  float csp = softplusf(sp_logc[0]) + 1e-4f;
  float ctm = softplusf(tm_logc[0]) + 1e-4f;
  float dsp = sd[((long)b * T_DIM + t) * CST_DIM + c] * __expf(sp_scale[s]);
  float dtm = td[((long)b * T_DIM + t) * CST_DIM + c] * __expf(tm_scale[s]);
  float xsp = log1pf(csp * dsp) / log1pf(csp * MAXSP);
  float xtm = log1pf(ctm * dtm) / log1pf(ctm * MAXTM);

  float acc = 0.f;
#pragma unroll
  for (int i = 0; i < FIREH; ++i) {
    float z = sp_w1[i] * xsp;
    acc += sp_w2[i] * (z / (1.f + __expf(-z)));   // silu
  }
#pragma unroll
  for (int i = 0; i < FIREH; ++i) {
    float z = tm_w1[i] * xtm;
    acc += tm_w2[i] * (z / (1.f + __expf(-z)));
  }
  out[idx] = acc;
}

// ---------------- generic GEMM: Y = X * W^T + bias ----------------
// X: (nrows, K) bf16 row-major (ld = K). W: (N, K) bf16 row-major (torch Linear layout).
// grid = (nrows/16, N/64); one wave per block computes a 16x64 strip:
// the A fragment is loaded once per k-step and reused across 4 WMMA N-tiles.
// mode 0: store bf16   mode 1: store f32   mode 2: add f32 (stream-ordered accumulation)
__global__ __launch_bounds__(32)
void gemm_wmma_k(const __bf16* __restrict__ X, const __bf16* __restrict__ W,
                 const float* __restrict__ bias, int K, void* __restrict__ Y,
                 int ldy, int mode) {
  int lane = threadIdx.x & 31;
  int l15 = lane & 15, hi = lane >> 4;
  long r0 = (long)blockIdx.x * 16;
  int  o0 = blockIdx.y * 64;
  const __bf16* xrow = X + (r0 + l15) * (long)K;
  const __bf16* wrow[4];
#pragma unroll
  for (int nt = 0; nt < 4; ++nt)
    wrow[nt] = W + (long)(o0 + nt * 16 + l15) * K;

  v8f c[4];
#pragma unroll
  for (int nt = 0; nt < 4; ++nt) {
    float bv = bias ? bias[o0 + nt * 16 + l15] : 0.f;
    c[nt] = (v8f){bv, bv, bv, bv, bv, bv, bv, bv};
  }

  for (int k0 = 0; k0 < K; k0 += 32) {
    __builtin_prefetch(xrow + k0 + 64, 0, 1);   // global_prefetch_b8 on the A stream
    v16bf a = fragA_global(xrow, k0, hi);
#pragma unroll
    for (int nt = 0; nt < 4; ++nt)
      c[nt] = wmma_bf16(a, fragB_rows(wrow[nt], k0, hi), c[nt]);
  }

  if (mode == 0) {
    __bf16* Yb = (__bf16*)Y;
#pragma unroll
    for (int nt = 0; nt < 4; ++nt)
#pragma unroll
      for (int j = 0; j < 8; ++j)
        Yb[(r0 + j + hi * 8) * (long)ldy + o0 + nt * 16 + l15] = (__bf16)c[nt][j];
  } else if (mode == 1) {
    float* Yf = (float*)Y;
#pragma unroll
    for (int nt = 0; nt < 4; ++nt)
#pragma unroll
      for (int j = 0; j < 8; ++j)
        Yf[(r0 + j + hi * 8) * (long)ldy + o0 + nt * 16 + l15] = c[nt][j];
  } else {
    float* Yf = (float*)Y;
#pragma unroll
    for (int nt = 0; nt < 4; ++nt)
#pragma unroll
      for (int j = 0; j < 8; ++j)
        Yf[(r0 + j + hi * 8) * (long)ldy + o0 + nt * 16 + l15] += c[nt][j];
  }
}

// ---------------- row self-attention over S (per b,t,head) ----------------
// Q/K/V/O all laid out (B,S,T,AH) bf16; row(s) = base + s*rstride.
// grid = (B*T*HALF_H, S/32), block = 64 (2 waves, each owning 16 query rows).
// P@V streams V through an 8KB LDS tile in 64-key chunks.
__global__ __launch_bounds__(64)
void row_attn_k(const __bf16* __restrict__ Q, const __bf16* __restrict__ Kb,
                const __bf16* __restrict__ Vb, __bf16* __restrict__ O) {
  __shared__ float  sc[2][16][S_DIM];          // 32 KB
  __shared__ __bf16 Vs[64 * D_DIM];            // 8 KB (one 64-key chunk of V)
  int wave = threadIdx.x >> 5, lane = threadIdx.x & 31;
  int tid = threadIdx.x;
  int l15 = lane & 15, hi = lane >> 4;
  int bx = blockIdx.x;
  int h = bx % HALF_H;
  int t = (bx / HALF_H) % T_DIM;
  int b = bx / (HALF_H * T_DIM);
  int q0 = blockIdx.y * 32 + wave * 16;

  const long rstride = (long)T_DIM * AH_DIM;
  const long base = ((long)b * S_DIM) * rstride + (long)t * AH_DIM + h * D_DIM;
  const __bf16* qrow = Q + base + (long)(q0 + l15) * rstride;

  // scores = (Q K^T) * scale  -> LDS
  for (int nt = 0; nt < S_DIM / 16; ++nt) {
    const __bf16* krow = Kb + base + (long)(nt * 16 + l15) * rstride;
    v8f c = {};
#pragma unroll
    for (int k0 = 0; k0 < D_DIM; k0 += 32)
      c = wmma_bf16(fragA_global(qrow, k0, hi), fragB_rows(krow, k0, hi), c);
#pragma unroll
    for (int j = 0; j < 8; ++j)
      sc[wave][j + hi * 8][nt * 16 + l15] = c[j] * SCALE_QK;
  }
  __syncthreads();

  // softmax over 256 keys: each lane owns half a row, combine across hi with shfl_xor(16)
  {
    int row = l15, cb = hi * 128;
    float m = -3.4e38f;
    for (int i = 0; i < 128; ++i) m = fmaxf(m, sc[wave][row][cb + i]);
    m = fmaxf(m, __shfl_xor(m, 16, 32));
    float s = 0.f;
    for (int i = 0; i < 128; ++i) {
      float e = __expf(sc[wave][row][cb + i] - m);
      sc[wave][row][cb + i] = e;
      s += e;
    }
    s += __shfl_xor(s, 16, 32);
    float inv = 1.f / s;
    for (int i = 0; i < 128; ++i) sc[wave][row][cb + i] *= inv;
  }

  // out = P @ V  (16 x 64), V streamed via LDS in 64-key chunks
  v8f acc[4] = {};
  for (int kc = 0; kc < S_DIM / 64; ++kc) {
    __syncthreads();   // previous chunk fully consumed / softmax done
    // cooperative stage: 64 keys x 64 dims, 16B vectors, 512 chunks over 64 threads
    for (int i = tid; i < 64 * (D_DIM / 8); i += 64) {
      int r = i >> 3, cblk = i & 7;
      *(v8bf*)&Vs[r * D_DIM + cblk * 8] =
          *(const v8bf*)(Vb + base + (long)(kc * 64 + r) * rstride + cblk * 8);
    }
    __syncthreads();
#pragma unroll
    for (int k0 = 0; k0 < 64; k0 += 32) {
      v16bf a = fragA_lds(&sc[wave][l15][0], kc * 64 + k0, hi);
#pragma unroll
      for (int nt = 0; nt < 4; ++nt)
        acc[nt] = wmma_bf16(a, fragB_lds_col(Vs, k0, hi, nt * 16 + l15), acc[nt]);
    }
  }
#pragma unroll
  for (int nt = 0; nt < 4; ++nt)
#pragma unroll
    for (int j = 0; j < 8; ++j)
      O[base + (long)(q0 + j + hi * 8) * rstride + nt * 16 + l15] = (__bf16)acc[nt][j];
}

// ---------------- ST cross attention (per b,s,head), keys = CST, with bias ----------------
// Q,O: (B,S,T,AH) ; K,V: (B,S,CST,AH). grid = B*S*HALF_H, block = 128 (4 waves cover T=64).
__global__ __launch_bounds__(128)
void st_attn_k(const __bf16* __restrict__ Q, const __bf16* __restrict__ Kb,
               const __bf16* __restrict__ Vb, const float* __restrict__ am,
               const float* __restrict__ fire, __bf16* __restrict__ O) {
  __shared__ float  sc[4][16][CST_DIM];        // 16 KB
  __shared__ __bf16 Vs[CST_DIM * D_DIM];       // 8 KB
  int wave = threadIdx.x >> 5, lane = threadIdx.x & 31;
  int tid = threadIdx.x;
  int l15 = lane & 15, hi = lane >> 4;
  int bx = blockIdx.x;
  int h = bx % HALF_H;
  int s = (bx / HALF_H) % S_DIM;
  int b = bx / (HALF_H * S_DIM);
  int q0 = wave * 16;

  const long qbase = (((long)b * S_DIM + s) * T_DIM) * AH_DIM + h * D_DIM;
  const long kbase = (((long)b * S_DIM + s) * CST_DIM) * AH_DIM + h * D_DIM;
  const long ambase = (long)b * T_DIM * CST_DIM;
  const long fbase = ((long)b * S_DIM + s) * T_DIM * CST_DIM;
  const __bf16* qrow = Q + qbase + (long)(q0 + l15) * AH_DIM;

  // stage V (64 x 64) while scores are computed
  for (int i = tid; i < CST_DIM * (D_DIM / 8); i += 128) {
    int r = i >> 3, cblk = i & 7;
    *(v8bf*)&Vs[r * D_DIM + cblk * 8] =
        *(const v8bf*)(Vb + kbase + (long)r * AH_DIM + cblk * 8);
  }

#pragma unroll
  for (int nt = 0; nt < CST_DIM / 16; ++nt) {
    const __bf16* krow = Kb + kbase + (long)(nt * 16 + l15) * AH_DIM;
    v8f c = {};
#pragma unroll
    for (int k0 = 0; k0 < D_DIM; k0 += 32)
      c = wmma_bf16(fragA_global(qrow, k0, hi), fragB_rows(krow, k0, hi), c);
#pragma unroll
    for (int j = 0; j < 8; ++j) {
      int tq = q0 + j + hi * 8;
      int cc = nt * 16 + l15;
      sc[wave][j + hi * 8][cc] = c[j] * SCALE_QK +
                                 am[ambase + (long)tq * CST_DIM + cc] +
                                 fire[fbase + (long)tq * CST_DIM + cc];
    }
  }
  __syncthreads();

  {  // softmax over 64 keys
    int row = l15, cb = hi * 32;
    float m = -3.4e38f;
    for (int i = 0; i < 32; ++i) m = fmaxf(m, sc[wave][row][cb + i]);
    m = fmaxf(m, __shfl_xor(m, 16, 32));
    float sum = 0.f;
    for (int i = 0; i < 32; ++i) {
      float e = __expf(sc[wave][row][cb + i] - m);
      sc[wave][row][cb + i] = e;
      sum += e;
    }
    sum += __shfl_xor(sum, 16, 32);
    float inv = 1.f / sum;
    for (int i = 0; i < 32; ++i) sc[wave][row][cb + i] *= inv;
  }
  __syncthreads();

  v8f acc[4] = {};
#pragma unroll
  for (int k0 = 0; k0 < CST_DIM; k0 += 32) {
    v16bf a = fragA_lds(&sc[wave][l15][0], k0, hi);
#pragma unroll
    for (int nt = 0; nt < 4; ++nt)
      acc[nt] = wmma_bf16(a, fragB_lds_col(Vs, k0, hi, nt * 16 + l15), acc[nt]);
  }
#pragma unroll
  for (int nt = 0; nt < 4; ++nt)
#pragma unroll
    for (int j = 0; j < 8; ++j)
      O[qbase + (long)(q0 + j + hi * 8) * AH_DIM + nt * 16 + l15] = (__bf16)acc[nt][j];
}

// ---------------- eco cross attention (per b,s,head), keys = CECO, K/V shared over s ----------------
// Q,O: (B,S,T,AH) ; K,V: (B,CECO,AH). grid = B*S*HALF_H, block = 128.
__global__ __launch_bounds__(128)
void eco_attn_k(const __bf16* __restrict__ Q, const __bf16* __restrict__ Kb,
                const __bf16* __restrict__ Vb, __bf16* __restrict__ O) {
  __shared__ float  sc[4][16][CECO_DIM];       // 8 KB
  __shared__ __bf16 Vs[CECO_DIM * D_DIM];      // 4 KB
  int wave = threadIdx.x >> 5, lane = threadIdx.x & 31;
  int tid = threadIdx.x;
  int l15 = lane & 15, hi = lane >> 4;
  int bx = blockIdx.x;
  int h = bx % HALF_H;
  int s = (bx / HALF_H) % S_DIM;
  int b = bx / (HALF_H * S_DIM);
  int q0 = wave * 16;

  const long qbase = (((long)b * S_DIM + s) * T_DIM) * AH_DIM + h * D_DIM;
  const long kbase = ((long)b * CECO_DIM) * AH_DIM + h * D_DIM;
  const __bf16* qrow = Q + qbase + (long)(q0 + l15) * AH_DIM;

  // stage V (32 x 64)
  for (int i = tid; i < CECO_DIM * (D_DIM / 8); i += 128) {
    int r = i >> 3, cblk = i & 7;
    *(v8bf*)&Vs[r * D_DIM + cblk * 8] =
        *(const v8bf*)(Vb + kbase + (long)r * AH_DIM + cblk * 8);
  }

#pragma unroll
  for (int nt = 0; nt < CECO_DIM / 16; ++nt) {
    const __bf16* krow = Kb + kbase + (long)(nt * 16 + l15) * AH_DIM;
    v8f c = {};
#pragma unroll
    for (int k0 = 0; k0 < D_DIM; k0 += 32)
      c = wmma_bf16(fragA_global(qrow, k0, hi), fragB_rows(krow, k0, hi), c);
#pragma unroll
    for (int j = 0; j < 8; ++j)
      sc[wave][j + hi * 8][nt * 16 + l15] = c[j] * SCALE_QK;
  }
  __syncthreads();

  {  // softmax over 32 keys
    int row = l15, cb = hi * 16;
    float m = -3.4e38f;
    for (int i = 0; i < 16; ++i) m = fmaxf(m, sc[wave][row][cb + i]);
    m = fmaxf(m, __shfl_xor(m, 16, 32));
    float sum = 0.f;
    for (int i = 0; i < 16; ++i) {
      float e = __expf(sc[wave][row][cb + i] - m);
      sc[wave][row][cb + i] = e;
      sum += e;
    }
    sum += __shfl_xor(sum, 16, 32);
    float inv = 1.f / sum;
    for (int i = 0; i < 16; ++i) sc[wave][row][cb + i] *= inv;
  }
  __syncthreads();

  v8f acc[4] = {};
  {  // single 32-wide K step
    v16bf a = fragA_lds(&sc[wave][l15][0], 0, hi);
#pragma unroll
    for (int nt = 0; nt < 4; ++nt)
      acc[nt] = wmma_bf16(a, fragB_lds_col(Vs, 0, hi, nt * 16 + l15), acc[nt]);
  }
#pragma unroll
  for (int nt = 0; nt < 4; ++nt)
#pragma unroll
    for (int j = 0; j < 8; ++j)
      O[qbase + (long)(q0 + j + hi * 8) * AH_DIM + nt * 16 + l15] = (__bf16)acc[nt][j];
}

// ---------------- host side ----------------
extern "C" void kernel_launch(void* const* d_in, const int* in_sizes, int n_in,
                              void* d_out, int out_size, void* d_ws, size_t ws_size,
                              hipStream_t stream) {
  (void)in_sizes; (void)n_in; (void)out_size; (void)ws_size;

  const float* hs  = (const float*)d_in[0];   // (B,S,T,H)
  const float* src = (const float*)d_in[1];   // (B,S,CST,H)
  const float* eco = (const float*)d_in[2];   // (B,CECO,H)
  const float* sd  = (const float*)d_in[3];   // (B,T,CST)
  const float* td  = (const float*)d_in[4];   // (B,T,CST)
  const float* am  = (const float*)d_in[5];   // (B,1,1,T,CST)
  // params flattened in dict order: [6..29] = {row,st,eco} x {q_w,q_b,k_w,k_b,v_w,v_b,o_w,o_b}
  // [30..37] = sp_log_c, sp_w1, sp_w2, tm_log_c, tm_w1, tm_w2, sp_scale, tm_scale
  const float* qkvo_w[12];
  const float* qkvo_b[12];
  for (int blk = 0; blk < 3; ++blk)
    for (int m = 0; m < 4; ++m) {
      qkvo_w[blk * 4 + m] = (const float*)d_in[6 + blk * 8 + m * 2];
      qkvo_b[blk * 4 + m] = (const float*)d_in[6 + blk * 8 + m * 2 + 1];
    }
  const float* sp_logc  = (const float*)d_in[30];
  const float* sp_w1    = (const float*)d_in[31];
  const float* sp_w2    = (const float*)d_in[32];
  const float* tm_logc  = (const float*)d_in[33];
  const float* tm_w1    = (const float*)d_in[34];
  const float* tm_w2    = (const float*)d_in[35];
  const float* sp_scale = (const float*)d_in[36];
  const float* tm_scale = (const float*)d_in[37];

  float* out = (float*)d_out;                 // (B,S,T,H) f32

  // workspace carve-up
  const long N_HS  = (long)B_DIM * S_DIM * T_DIM;      // 65536 rows
  const long N_SRC = (long)B_DIM * S_DIM * CST_DIM;    // 65536 rows
  const long N_ECO = (long)B_DIM * CECO_DIM;           // 128 rows
  const long HS_E  = N_HS * H_DIM;                     // 33.5M
  const long SRC_E = N_SRC * H_DIM;
  const long ECO_E = N_ECO * H_DIM;
  const long W_E   = (long)AH_DIM * H_DIM;             // 131072 per weight matrix
  const long QKV_E = N_HS * AH_DIM;                    // 16.7M
  const long FIRE_E = (long)B_DIM * S_DIM * T_DIM * CST_DIM;

  char* ws = (char*)d_ws;
  size_t off = 0;
  auto take = [&](long elems, int esz) { char* p = ws + off; off += ((size_t)elems * esz + 255) & ~(size_t)255; return p; };
  __bf16* hsb  = (__bf16*)take(HS_E, 2);
  __bf16* srcb = (__bf16*)take(SRC_E, 2);
  __bf16* ecob = (__bf16*)take(ECO_E, 2);
  __bf16* wb   = (__bf16*)take(12 * W_E, 2);
  __bf16* bufQ = (__bf16*)take(QKV_E, 2);
  __bf16* bufK = (__bf16*)take(QKV_E, 2);
  __bf16* bufV = (__bf16*)take(QKV_E, 2);
  __bf16* bufO = (__bf16*)take(QKV_E, 2);
  float*  fire = (float*)take(FIRE_E, 4);

  // 1. casts
  cast_bf16_k<<<4096, 256, 0, stream>>>(hs,  hsb,  HS_E);
  cast_bf16_k<<<4096, 256, 0, stream>>>(src, srcb, SRC_E);
  cast_bf16_k<<<256, 256, 0, stream>>>(eco, ecob, ECO_E);
  for (int i = 0; i < 12; ++i)
    cast_bf16_k<<<512, 256, 0, stream>>>(qkvo_w[i], wb + (long)i * W_E, W_E);

  // 2. FIRE bias
  fire_bias_k<<<(FIRE_E + 255) / 256, 256, 0, stream>>>(
      sd, td, sp_logc, sp_w1, sp_w2, tm_logc, tm_w1, tm_w2, sp_scale, tm_scale, fire);

  dim3 gProj(N_HS / 16, AH_DIM / 64);    // (4096, 4)  : 16x64 strip per wave
  dim3 gOut(N_HS / 16, H_DIM / 64);      // (4096, 8)
  dim3 gEco(N_ECO / 16, AH_DIM / 64);    // (8, 4)

  // 3. row self-attention block
  gemm_wmma_k<<<gProj, 32, 0, stream>>>(hsb, wb + 0 * W_E, qkvo_b[0], H_DIM, bufQ, AH_DIM, 0);
  gemm_wmma_k<<<gProj, 32, 0, stream>>>(hsb, wb + 1 * W_E, qkvo_b[1], H_DIM, bufK, AH_DIM, 0);
  gemm_wmma_k<<<gProj, 32, 0, stream>>>(hsb, wb + 2 * W_E, qkvo_b[2], H_DIM, bufV, AH_DIM, 0);
  row_attn_k<<<dim3(B_DIM * T_DIM * HALF_H, S_DIM / 32), 64, 0, stream>>>(bufQ, bufK, bufV, bufO);
  gemm_wmma_k<<<gOut, 32, 0, stream>>>(bufO, wb + 3 * W_E, qkvo_b[3], AH_DIM, out, H_DIM, 1);

  // 4. ST cross-attention block
  gemm_wmma_k<<<gProj, 32, 0, stream>>>(hsb,  wb + 4 * W_E, qkvo_b[4], H_DIM, bufQ, AH_DIM, 0);
  gemm_wmma_k<<<gProj, 32, 0, stream>>>(srcb, wb + 5 * W_E, qkvo_b[5], H_DIM, bufK, AH_DIM, 0);
  gemm_wmma_k<<<gProj, 32, 0, stream>>>(srcb, wb + 6 * W_E, qkvo_b[6], H_DIM, bufV, AH_DIM, 0);
  st_attn_k<<<B_DIM * S_DIM * HALF_H, 128, 0, stream>>>(bufQ, bufK, bufV, am, fire, bufO);
  gemm_wmma_k<<<gOut, 32, 0, stream>>>(bufO, wb + 7 * W_E, qkvo_b[7], AH_DIM, out, H_DIM, 2);

  // 5. eco cross-attention block
  gemm_wmma_k<<<gProj, 32, 0, stream>>>(hsb,  wb + 8 * W_E, qkvo_b[8],  H_DIM, bufQ, AH_DIM, 0);
  gemm_wmma_k<<<gEco, 32, 0, stream>>>(ecob, wb + 9 * W_E, qkvo_b[9],  H_DIM, bufK, AH_DIM, 0);
  gemm_wmma_k<<<gEco, 32, 0, stream>>>(ecob, wb + 10 * W_E, qkvo_b[10], H_DIM, bufV, AH_DIM, 0);
  eco_attn_k<<<B_DIM * S_DIM * HALF_H, 128, 0, stream>>>(bufQ, bufK, bufV, bufO);
  gemm_wmma_k<<<gOut, 32, 0, stream>>>(bufO, wb + 11 * W_E, qkvo_b[11], AH_DIM, out, H_DIM, 2);
}
